// SelectiveSSMLayer_60833916780826
// MI455X (gfx1250) — matrix operand order repaired and checked by reading, not compiled, not run on hardware
//
#include <hip/hip_runtime.h>

typedef __bf16 bf16_t;
typedef __attribute__((ext_vector_type(16))) __bf16 v16bf;
typedef __attribute__((ext_vector_type(8)))  __bf16 bf16x8;
typedef __attribute__((ext_vector_type(8)))  float  v8f;

#define B_SZ  2048
#define S_SZ  512
#define IN_SZ 512
#define R_SZ  32
#define H_SZ  2048
#define CC_SZ 1536   // 2S + IN

static constexpr float DT_C     = 0.01f;
static constexpr float MARGIN_C = 0.01f;
static constexpr float LN_EPS_C = 1e-5f;

// ---------------------------------------------------------------------------
// f32 -> bf16 conversion (weights)
// ---------------------------------------------------------------------------
__global__ void cvt_f32_to_bf16(const float* __restrict__ in,
                                bf16_t* __restrict__ out, long long n) {
  long long i = (long long)blockIdx.x * blockDim.x + threadIdx.x;
  if (i < n) out[i] = (bf16_t)in[i];
}

// Build concat activation buffer cc[b, 0:512]=h, cc[b,512:1024]=u (bf16).
// Columns 1024:1536 are filled later by the selection GEMM (z).
__global__ void pack_cc(const float* __restrict__ h, const float* __restrict__ u,
                        bf16_t* __restrict__ cc) {
  int idx = blockIdx.x * blockDim.x + threadIdx.x;  // B * 1024 threads
  int b = idx >> 10;
  int c = idx & 1023;
  float v = (c < S_SZ) ? h[(size_t)b * S_SZ + c] : u[(size_t)b * IN_SZ + (c - S_SZ)];
  cc[(size_t)b * CC_SZ + c] = (bf16_t)v;
}

// ---------------------------------------------------------------------------
// CDNA5 async global->LDS helpers (ASYNCcnt-tracked DMA writes into LDS).
// INST_OFFSET is added to BOTH the LDS and global address, so one base pair
// covers 32 contiguous bytes with offset:0 / offset:16.
// Low 32 bits of a generic LDS pointer == LDS byte address (aperture rule).
// ---------------------------------------------------------------------------
__device__ __forceinline__ void async_tile_load(unsigned ldsA, unsigned long long gA,
                                                unsigned ldsW, unsigned long long gW) {
  asm volatile(
      "global_load_async_to_lds_b128 %0, %2, off\n\t"
      "global_load_async_to_lds_b128 %0, %2, off offset:16\n\t"
      "global_load_async_to_lds_b128 %1, %3, off\n\t"
      "global_load_async_to_lds_b128 %1, %3, off offset:16"
      :
      : "v"(ldsA), "v"(ldsW), "v"(gA), "v"(gW)
      : "memory");
}

__device__ __forceinline__ void wait_async_le4() {
  asm volatile("s_wait_asynccnt 0x4" ::: "memory");
}
__device__ __forceinline__ void wait_async_0() {
  asm volatile("s_wait_asynccnt 0x0" ::: "memory");
}

// ---------------------------------------------------------------------------
// bf16 WMMA GEMM:  C[m,n] = act( sum_k A[m,k] * W[n,k] + bias[n] )
//   A: (M x K) row-major, leading dim lda (bf16)
//   W: (N x K) row-major, leading dim ldw (bf16)  -> used as B^T
// Output: optional f32 buffer Cf (ldcf) and/or bf16 buffer Cb (ldcb).
// act: 0=none, 1=SiLU, 2=sigmoid
// 256 threads (8 waves), tile 128x128, BK=32, double-buffered async LDS.
// ---------------------------------------------------------------------------
#define BM 128
#define BN 128
#define BK 32
#define LDSL 40   // padded LDS row length in bf16 elems (80 B, 16 B aligned)

__global__ __launch_bounds__(256) void gemm_bf16_wmma(
    const bf16_t* __restrict__ A, int lda,
    const bf16_t* __restrict__ W, int ldw,
    const float* __restrict__ bias,
    float* __restrict__ Cf, int ldcf,
    bf16_t* __restrict__ Cb, int ldcb,
    int K, int act) {
  __shared__ bf16_t sA[2][BM * LDSL];
  __shared__ bf16_t sW[2][BN * LDSL];

  const int tid  = threadIdx.x;
  const int lane = tid & 31;
  const int wave = tid >> 5;
  const int wm   = wave & 1;   // 2 waves along M
  const int wn   = wave >> 1;  // 4 waves along N
  const int bM   = blockIdx.y * BM;
  const int bN   = blockIdx.x * BN;

  v8f acc[4][2];
  const v8f vzero = {0.f, 0.f, 0.f, 0.f, 0.f, 0.f, 0.f, 0.f};
#pragma unroll
  for (int i = 0; i < 4; ++i)
#pragma unroll
    for (int j = 0; j < 2; ++j) acc[i][j] = vzero;

  const int lrow = tid >> 1;        // 0..127 (tile row this thread loads)
  const int lcol = (tid & 1) * 16;  // 0 or 16 (16 bf16 = 32 B per thread)
  const int half = lane >> 4;       // 0 or 1
  const int l15  = lane & 15;

  // Per-thread async staging addresses
  const bf16_t* gA = A + (size_t)(bM + lrow) * lda + lcol;
  const bf16_t* gW = W + (size_t)(bN + lrow) * ldw + lcol;
  unsigned ldsA[2], ldsW[2];
#pragma unroll
  for (int b2 = 0; b2 < 2; ++b2) {
    ldsA[b2] = (unsigned)(size_t)&sA[b2][lrow * LDSL + lcol];
    ldsW[b2] = (unsigned)(size_t)&sW[b2][lrow * LDSL + lcol];
  }

  const int nTiles = K / BK;
  // Prologue: kick off tile 0 into buffer 0
  async_tile_load(ldsA[0], (unsigned long long)gA,
                  ldsW[0], (unsigned long long)gW);

  for (int t = 0; t < nTiles; ++t) {
    const int cur = t & 1;
    if (t + 1 < nTiles) {
      // Prefetch next tile into the other buffer (safe: end-of-iteration
      // barrier below guarantees no wave still reads it).
      async_tile_load(ldsA[cur ^ 1], (unsigned long long)(gA + (size_t)(t + 1) * BK),
                      ldsW[cur ^ 1], (unsigned long long)(gW + (size_t)(t + 1) * BK));
      wait_async_le4();  // async loads complete in order: tile t has landed
    } else {
      wait_async_0();
    }
    __syncthreads();  // all lanes' LDS writes for tile t visible

    const bf16_t* tA = sA[cur];
    const bf16_t* tW = sW[cur];

    // B fragments (32x16, K x N): lane n=l15; VGPR0-3 hold K = half*8+0..7,
    // VGPR4-7 hold K = 16 + half*8 + 0..7 (same order as A).
    v16bf bfrag[2];
#pragma unroll
    for (int ni = 0; ni < 2; ++ni) {
      int n = wn * 32 + ni * 16 + l15;
      const bf16_t* p = &tW[n * LDSL + half * 8];
      bf16x8 lo = *(const bf16x8*)p;
      bf16x8 hi = *(const bf16x8*)(p + 16);
      bfrag[ni] = __builtin_shufflevector(lo, hi, 0, 1, 2, 3, 4, 5, 6, 7, 8, 9,
                                          10, 11, 12, 13, 14, 15);
    }
    // A fragments (16x32): lane m=l15; VGPR0-3 hold K = half*8+0..7,
    // VGPR4-7 hold K = 16 + half*8 + 0..7.
#pragma unroll
    for (int mi = 0; mi < 4; ++mi) {
      int m = wm * 64 + mi * 16 + l15;
      const bf16_t* p = &tA[m * LDSL + half * 8];
      bf16x8 lo = *(const bf16x8*)p;         // K = half*8 .. +7
      bf16x8 hi = *(const bf16x8*)(p + 16);  // K = 16 + half*8 .. +7
      v16bf afrag = __builtin_shufflevector(lo, hi, 0, 1, 2, 3, 4, 5, 6, 7, 8,
                                            9, 10, 11, 12, 13, 14, 15);
#pragma unroll
      for (int ni = 0; ni < 2; ++ni)
        acc[mi][ni] = __builtin_amdgcn_wmma_f32_16x16x32_bf16(
            false, afrag, false, bfrag[ni], (short)0, acc[mi][ni], false, false);
    }
    __syncthreads();  // all reads of buffer `cur` done before it is refilled
  }

  // Epilogue. C/D layout: elem e of v8f -> row (half*8 + e), col l15.
#pragma unroll
  for (int mi = 0; mi < 4; ++mi) {
#pragma unroll
    for (int ni = 0; ni < 2; ++ni) {
      int nG = bN + wn * 32 + ni * 16 + l15;
      float bv = bias ? bias[nG] : 0.0f;
#pragma unroll
      for (int e = 0; e < 8; ++e) {
        int mG = bM + wm * 64 + mi * 16 + half * 8 + e;
        float v = acc[mi][ni][e] + bv;
        if (act == 1) {
          v = v / (1.0f + __expf(-v));  // SiLU
        } else if (act == 2) {
          v = 1.0f / (1.0f + __expf(-v));  // sigmoid
        }
        if (Cf) Cf[(size_t)mG * ldcf + nG] = v;
        if (Cb) Cb[(size_t)mG * ldcb + nG] = (bf16_t)v;
      }
    }
  }
}

// ---------------------------------------------------------------------------
// Vh[b,r] = sum_d V[d,r] * h[b,d]   (tiny: 2048 x 32, K=512)
// ---------------------------------------------------------------------------
__global__ __launch_bounds__(256) void vh_compute(const float* __restrict__ h,
                                                  const float* __restrict__ V,
                                                  float* __restrict__ Vh) {
  __shared__ float red[256];
  int b = blockIdx.x;
  int t = threadIdx.x;
  int r = t & 31;
  int seg = t >> 5;  // 8 segments of 64 along d
  float s = 0.f;
  const float* hb = h + (size_t)b * S_SZ;
#pragma unroll 4
  for (int d = seg * 64; d < seg * 64 + 64; ++d) s += V[(size_t)d * R_SZ + r] * hb[d];
  red[t] = s;
  __syncthreads();
  if (t < 32) {
    float tot = 0.f;
#pragma unroll
    for (int i = 0; i < 8; ++i) tot += red[i * 32 + t];
    Vh[(size_t)b * R_SZ + t] = tot;
  }
}

// ---------------------------------------------------------------------------
// Fused SSM update + residual + LayerNorm. One block per batch row.
// ---------------------------------------------------------------------------
__global__ __launch_bounds__(256) void ssm_finalize(
    const float* __restrict__ h, const float* __restrict__ zf,
    const float* __restrict__ dd, const bf16_t* __restrict__ du,
    const float* __restrict__ bu, const float* __restrict__ vhv,
    const float* __restrict__ D_base, const float* __restrict__ U_base,
    const float* __restrict__ ln_g, const float* __restrict__ ln_b,
    float* __restrict__ out) {
  __shared__ float vsh[R_SZ];
  __shared__ float ysh[S_SZ];
  __shared__ float red[256];
  int b = blockIdx.x;
  int t = threadIdx.x;
  if (t < R_SZ) vsh[t] = vhv[(size_t)b * R_SZ + t];
  __syncthreads();

  float ly[2];
#pragma unroll
  for (int j = 0; j < 2; ++j) {
    int d = t + j * 256;
    size_t bd = (size_t)b * S_SZ + d;
    float hv  = h[bd];
    float z   = zf[bd];
    float ddv = dd[bd];
    // D_t = -softplus(D_base + delta_d) - margin  (numerically stable)
    float x  = D_base[d] + ddv;
    float sp = fmaxf(x, 0.f) + log1pf(__expf(-fabsf(x)));
    float eD = __expf((-sp - MARGIN_C) * DT_C);
    // UVh = z * sum_r (U_base[d,r] + delta_u[b,d,r]) * Vh[b,r]
    const bf16_t* dup = du + bd * R_SZ;
    const float* ubp  = U_base + (size_t)d * R_SZ;
    float uvh = 0.f;
#pragma unroll
    for (int r = 0; r < R_SZ; ++r) uvh += (ubp[r] + (float)dup[r]) * vsh[r];
    uvh *= z;
    float h_next = eD * (hv + uvh * DT_C) + bu[bd] * z * DT_C;
    float y = h_next + hv;
    ysh[d] = y;
    ly[j]  = y;
  }

  // mean
  red[t] = ly[0] + ly[1];
  __syncthreads();
  for (int off = 128; off > 0; off >>= 1) {
    if (t < off) red[t] += red[t + off];
    __syncthreads();
  }
  float mu = red[0] * (1.0f / S_SZ);
  __syncthreads();
  // variance
  float d0 = ly[0] - mu, d1 = ly[1] - mu;
  red[t] = d0 * d0 + d1 * d1;
  __syncthreads();
  for (int off = 128; off > 0; off >>= 1) {
    if (t < off) red[t] += red[t + off];
    __syncthreads();
  }
  float inv = rsqrtf(red[0] * (1.0f / S_SZ) + LN_EPS_C);

#pragma unroll
  for (int j = 0; j < 2; ++j) {
    int d = t + j * 256;
    out[(size_t)b * S_SZ + d] = (ysh[d] - mu) * inv * ln_g[d] + ln_b[d];
  }
}

// ---------------------------------------------------------------------------
// Host-side orchestration
// ---------------------------------------------------------------------------
extern "C" void kernel_launch(void* const* d_in, const int* in_sizes, int n_in,
                              void* d_out, int out_size, void* d_ws, size_t ws_size,
                              hipStream_t stream) {
  (void)in_sizes; (void)n_in; (void)out_size; (void)ws_size;
  const float* h      = (const float*)d_in[0];
  const float* u      = (const float*)d_in[1];
  const float* sel_W1 = (const float*)d_in[2];
  const float* sel_b1 = (const float*)d_in[3];
  const float* sel_W2 = (const float*)d_in[4];
  const float* sel_b2 = (const float*)d_in[5];
  const float* dD_W1  = (const float*)d_in[6];
  const float* dD_b1  = (const float*)d_in[7];
  const float* dD_W2  = (const float*)d_in[8];
  const float* dD_b2  = (const float*)d_in[9];
  const float* dU_W1  = (const float*)d_in[10];
  const float* dU_b1  = (const float*)d_in[11];
  const float* dU_W2  = (const float*)d_in[12];
  const float* dU_b2  = (const float*)d_in[13];
  const float* D_base = (const float*)d_in[14];
  const float* U_base = (const float*)d_in[15];
  const float* V      = (const float*)d_in[16];
  const float* B_w    = (const float*)d_in[17];
  const float* ln_g   = (const float*)d_in[18];
  const float* ln_b   = (const float*)d_in[19];

  char* base = (char*)d_ws;
  size_t off = 0;
  auto carve = [&](size_t bytes) -> void* {
    void* p = base + off;
    off = (off + bytes + 255) & ~(size_t)255;
    return p;
  };
  bf16_t* cc  = (bf16_t*)carve((size_t)B_SZ * CC_SZ * 2);          // [h,u,z] bf16
  bf16_t* hid = (bf16_t*)carve((size_t)B_SZ * H_SZ * 2);           // reused hidden
  bf16_t* w1  = (bf16_t*)carve((size_t)H_SZ * (S_SZ + IN_SZ) * 2); // sel_W1
  bf16_t* w2  = (bf16_t*)carve((size_t)S_SZ * H_SZ * 2);           // sel_W2
  bf16_t* wd1 = (bf16_t*)carve((size_t)H_SZ * CC_SZ * 2);          // dD_W1
  bf16_t* wd2 = (bf16_t*)carve((size_t)S_SZ * H_SZ * 2);           // dD_W2
  bf16_t* wu1 = (bf16_t*)carve((size_t)H_SZ * CC_SZ * 2);          // dU_W1
  bf16_t* wu2 = (bf16_t*)carve((size_t)S_SZ * R_SZ * H_SZ * 2);    // dU_W2
  bf16_t* wb  = (bf16_t*)carve((size_t)S_SZ * IN_SZ * 2);          // B_w
  float*  zf  = (float*)carve((size_t)B_SZ * S_SZ * 4);            // z (f32)
  float*  dd  = (float*)carve((size_t)B_SZ * S_SZ * 4);            // delta_d
  bf16_t* du  = (bf16_t*)carve((size_t)B_SZ * S_SZ * R_SZ * 2);    // delta_u bf16
  float*  bu  = (float*)carve((size_t)B_SZ * S_SZ * 4);            // u @ B_w^T
  float*  vh  = (float*)carve((size_t)B_SZ * R_SZ * 4);            // V^T h

  auto cvt = [&](const float* in, bf16_t* out, long long n) {
    cvt_f32_to_bf16<<<(unsigned)((n + 255) / 256), 256, 0, stream>>>(in, out, n);
  };
  cvt(sel_W1, w1, (long long)H_SZ * (S_SZ + IN_SZ));
  cvt(sel_W2, w2, (long long)S_SZ * H_SZ);
  cvt(dD_W1, wd1, (long long)H_SZ * CC_SZ);
  cvt(dD_W2, wd2, (long long)S_SZ * H_SZ);
  cvt(dU_W1, wu1, (long long)H_SZ * CC_SZ);
  cvt(dU_W2, wu2, (long long)S_SZ * R_SZ * H_SZ);
  cvt(B_w, wb, (long long)S_SZ * IN_SZ);
  pack_cc<<<(B_SZ * 1024) / 256, 256, 0, stream>>>(h, u, cc);

  const dim3 blk(256);
  const int gy = B_SZ / BM;  // 16
  // sel L1: H1 = silu(x @ sel_W1^T + b)     (K=1024, N=2048) -> hid (bf16)
  gemm_bf16_wmma<<<dim3(H_SZ / BN, gy), blk, 0, stream>>>(
      cc, CC_SZ, w1, S_SZ + IN_SZ, sel_b1, nullptr, 0, hid, H_SZ, S_SZ + IN_SZ, 1);
  // sel L2: z = sigmoid(H1 @ sel_W2^T + b)  (K=2048, N=512) -> zf (f32) + cc[:,1024:]
  gemm_bf16_wmma<<<dim3(S_SZ / BN, gy), blk, 0, stream>>>(
      hid, H_SZ, w2, H_SZ, sel_b2, zf, S_SZ, cc + 1024, CC_SZ, H_SZ, 2);
  // dD L1: HD = silu(cc @ dD_W1^T + b)      (K=1536, N=2048) -> hid
  gemm_bf16_wmma<<<dim3(H_SZ / BN, gy), blk, 0, stream>>>(
      cc, CC_SZ, wd1, CC_SZ, dD_b1, nullptr, 0, hid, H_SZ, CC_SZ, 1);
  // dD L2: delta_d = HD @ dD_W2^T + b       (K=2048, N=512) -> dd (f32)
  gemm_bf16_wmma<<<dim3(S_SZ / BN, gy), blk, 0, stream>>>(
      hid, H_SZ, wd2, H_SZ, dD_b2, dd, S_SZ, nullptr, 0, H_SZ, 0);
  // dU L1: HU = silu(cc @ dU_W1^T + b)      (K=1536, N=2048) -> hid
  gemm_bf16_wmma<<<dim3(H_SZ / BN, gy), blk, 0, stream>>>(
      cc, CC_SZ, wu1, CC_SZ, dU_b1, nullptr, 0, hid, H_SZ, CC_SZ, 1);
  // dU L2: delta_u = HU @ dU_W2^T + b       (K=2048, N=16384) -> du (bf16)
  gemm_bf16_wmma<<<dim3((S_SZ * R_SZ) / BN, gy), blk, 0, stream>>>(
      hid, H_SZ, wu2, H_SZ, dU_b2, nullptr, 0, du, S_SZ * R_SZ, H_SZ, 0);
  // Bu = u @ B_w^T                          (K=512, N=512) -> bu (f32)
  gemm_bf16_wmma<<<dim3(S_SZ / BN, gy), blk, 0, stream>>>(
      cc + S_SZ, CC_SZ, wb, IN_SZ, nullptr, bu, S_SZ, nullptr, 0, IN_SZ, 0);

  vh_compute<<<B_SZ, 256, 0, stream>>>(h, V, vh);
  ssm_finalize<<<B_SZ, 256, 0, stream>>>(h, zf, dd, du, bu, vh, D_base, U_base,
                                         ln_g, ln_b, (float*)d_out);
}
